// GlobalGraph_76450417869352
// MI455X (gfx1250) — compile-verified, matrix operand order checked by reading, not run on hardware
//
#include <hip/hip_runtime.h>
#include <stdint.h>

// ---------------------------------------------------------------------------
// Types for WMMA bf16 path (gfx1250, wave32)
// ---------------------------------------------------------------------------
typedef __attribute__((ext_vector_type(16))) __bf16 bf16x16;
typedef __attribute__((ext_vector_type(8)))  float  f32x8;
typedef __attribute__((ext_vector_type(4)))  int    i32x4;

union Frag16 {
    bf16x16  v;
    uint32_t u[8];
};

#if __has_builtin(__builtin_amdgcn_global_load_async_to_lds_b128)
#define HAVE_ASYNC_LDS 1
#endif

__device__ __forceinline__ void wait_asynccnt0() {
#if defined(HAVE_ASYNC_LDS)
#if __has_builtin(__builtin_amdgcn_s_wait_asynccnt)
    __builtin_amdgcn_s_wait_asynccnt(0);
#else
    asm volatile("s_wait_asynccnt 0x0" ::: "memory");
#endif
#endif
}

__device__ __forceinline__ uint16_t f2bf(float f) {
    uint32_t b = __float_as_uint(f);
    uint32_t r = b + 0x7FFFu + ((b >> 16) & 1u);   // round-to-nearest-even
    return (uint16_t)(r >> 16);
}

__device__ __forceinline__ unsigned enc_f32(float f) {
    unsigned u = __float_as_uint(f);
    return (f >= 0.0f) ? (u | 0x80000000u) : ~u;   // order-preserving encode
}
__device__ __forceinline__ float dec_f32(unsigned u) {
    return (u & 0x80000000u) ? __uint_as_float(u & 0x7FFFFFFFu)
                             : __uint_as_float(~u);
}

// ---------------------------------------------------------------------------
// Weight prep: Bt[n*K + k] = bf16(B[k*N + n])  (done once; B is 256x256)
// ---------------------------------------------------------------------------
__global__ void to_bf16_t(const float* __restrict__ B, uint16_t* __restrict__ Bt,
                          int K, int N) {
    int idx = blockIdx.x * 256 + threadIdx.x;
    if (idx >= K * N) return;
    int k = idx / N, n = idx % N;                  // coalesced read along n
    Bt[(size_t)n * K + k] = f2bf(B[idx]);
}

// ---------------------------------------------------------------------------
// Dual-input WMMA GEMM:  C = A0@B0t (+ A1@B1t) (+ bias)
// A fp32 row-major [M][K]; Bt bf16 "transposed" [N][K]; C fp32 [M][N].
// Block: 256 threads (8 waves). Block tile 128(M) x 64(N), K stepped by 32.
// Wave w computes rows [m0+16w, m0+16w+16), all 64 columns (4 wmma accums).
// ---------------------------------------------------------------------------
__global__ __launch_bounds__(256)
void gemm_dual_bf16_wmma(const float* __restrict__ A0, const uint16_t* __restrict__ B0,
                         const float* __restrict__ A1, const uint16_t* __restrict__ B1,
                         const float* __restrict__ bias, float* __restrict__ C,
                         int M, int N, int K) {
    __shared__ uint32_t ldsA[128 * 16];   // row-major  [m][k-pairs]  8 KB
    __shared__ uint16_t ldsB[64 * 32];    // "col-major" [n][k]       4 KB

    const int tid  = threadIdx.x;
    const int lane = tid & 31;
    const int wave = tid >> 5;
    const int half = lane >> 4;     // 0: lanes 0-15, 1: lanes 16-31
    const int l16  = lane & 15;
    const int m0   = blockIdx.x * 128;
    const int n0   = blockIdx.y * 64;

    f32x8 acc[4] = {};

    for (int pass = 0; pass < 2; ++pass) {
        const float*    A = pass ? A1 : A0;
        const uint16_t* B = pass ? B1 : B0;
        if (A == nullptr) break;                       // uniform

        for (int k0 = 0; k0 < K; k0 += 32) {
            // ---- B tile 64(n) x 32(k) bf16: straight copy from Bt ----
#if defined(HAVE_ASYNC_LDS)
            {   // one async b128 per thread (4096 B total)
                int nn = tid >> 2, c8 = tid & 3;       // 8 bf16 per chunk
                const uint16_t* gp = B + (size_t)(n0 + nn) * K + k0 + 8 * c8;
                uint16_t* lp = &ldsB[nn * 32 + 8 * c8];
                __builtin_amdgcn_global_load_async_to_lds_b128(
                    (i32x4*)gp, (i32x4*)lp, 0, 0);
            }
#else
            #pragma unroll
            for (int i = 0; i < 2; ++i) {              // uint2 = 4 bf16
                int idx = tid + i * 256;               // 0..511
                int nn = idx >> 3, c4 = idx & 7;
                uint2 v = *(const uint2*)(B + (size_t)(n0 + nn) * K + k0 + 4 * c4);
                ((uint2*)ldsB)[nn * 8 + c4] = v;
            }
#endif
            // ---- A tile 128(m) x 32(k): fp32 -> bf16, float4 loads ----
            #pragma unroll
            for (int i = 0; i < 4; ++i) {
                int idx = tid + i * 256;               // 0..1023
                int r = idx >> 3, c4 = idx & 7;        // 8 float4 per row
                int gr = m0 + r;
                gr = (gr < M) ? gr : (M - 1);          // clamp: rows >= M never stored
                float4 f = *(const float4*)(A + (size_t)gr * K + k0 + 4 * c4);
                uint32_t lo = (uint32_t)f2bf(f.x) | ((uint32_t)f2bf(f.y) << 16);
                uint32_t hi = (uint32_t)f2bf(f.z) | ((uint32_t)f2bf(f.w) << 16);
                ((uint2*)ldsA)[r * 8 + c4] = make_uint2(lo, hi);
            }
            wait_asynccnt0();
            __syncthreads();

            // A fragment: 16x32 bf16 per ISA layout (lanes<16: M=l16, K pairs
            // 0..7 & 16..23; lanes>=16: K pairs 8..15 & 24..31)
            const uint32_t* Au = ldsA;                   // 16 dwords per row
            const uint32_t* Bu = (const uint32_t*)ldsB;  // 16 dwords per col
            Frag16 a;
            #pragma unroll
            for (int j = 0; j < 8; ++j) {
                int kb = ((j & 4) << 2) + half * 8 + ((j & 3) << 1);
                a.u[j] = Au[(wave * 16 + l16) * 16 + (kb >> 1)];
            }
            #pragma unroll
            for (int t = 0; t < 4; ++t) {
                Frag16 b;
                #pragma unroll
                for (int j = 0; j < 8; ++j) {
                    int kb = ((j & 4) << 2) + half * 8 + ((j & 3) << 1);
                    b.u[j] = Bu[(t * 16 + l16) * 16 + (kb >> 1)];
                }
                acc[t] = __builtin_amdgcn_wmma_f32_16x16x32_bf16(
                    false, a.v, false, b.v, (short)0, acc[t], false, false);
            }
            __syncthreads();
        }
    }

    // Epilogue: C/D layout — vgpr r holds M = r + 8*half, N = l16 within tile
    #pragma unroll
    for (int t = 0; t < 4; ++t) {
        int col = n0 + t * 16 + l16;
        float bv = bias ? bias[col] : 0.0f;
        #pragma unroll
        for (int r = 0; r < 8; ++r) {
            int row = m0 + wave * 16 + half * 8 + r;
            if (row < M) C[(size_t)row * N + col] = acc[t][r] + bv;
        }
    }
}

// ---------------------------------------------------------------------------
// Utility: zero a region (also used for m_enc since 0 encodes -inf sentinel)
// ---------------------------------------------------------------------------
__global__ void zero_f32(float* p, size_t n) {
    size_t i = (size_t)blockIdx.x * 256 + threadIdx.x;
    if (i < n) p[i] = 0.0f;
}

// ---------------------------------------------------------------------------
// al_s[i] = dot(h[i], att_src); al_d[i] = dot(h[i], att_dst). 1 wave / row.
// ---------------------------------------------------------------------------
__global__ __launch_bounds__(256)
void rowdots(const float* __restrict__ h, const float* __restrict__ as,
             const float* __restrict__ ad, float* __restrict__ als,
             float* __restrict__ ald, int Nn) {
    int row  = blockIdx.x * 8 + (threadIdx.x >> 5);
    int lane = threadIdx.x & 31;
    if (row >= Nn) return;
    const float* hp = h + (size_t)row * 256 + lane * 8;
    float s = 0.0f, d = 0.0f;
    #pragma unroll
    for (int i = 0; i < 8; ++i) {
        float v = hp[i];
        s += v * as[lane * 8 + i];
        d += v * ad[lane * 8 + i];
    }
    #pragma unroll
    for (int off = 16; off > 0; off >>= 1) {
        s += __shfl_down(s, off, 32);
        d += __shfl_down(d, off, 32);
    }
    if (lane == 0) { als[row] = s; ald[row] = d; }
}

// Items 0..E-1 are edges, items E..E+Nn-1 are self loops (s=d=item-E).
__device__ __forceinline__ void gat_item(int item, int E, const int* src,
                                         const int* dst, int& s, int& d) {
    if (item < E) { s = src[item]; d = dst[item]; }
    else          { s = d = item - E; }
}

__global__ __launch_bounds__(256)
void gat_segmax(const int* __restrict__ src, const int* __restrict__ dst, int E,
                int Nn, const float* __restrict__ als,
                const float* __restrict__ ald, unsigned* __restrict__ menc) {
    int item = blockIdx.x * 256 + threadIdx.x;
    if (item >= E + Nn) return;
    int s, d; gat_item(item, E, src, dst, s, d);
    float e = als[s] + ald[d];
    e = (e > 0.0f) ? e : 0.2f * e;                    // leaky_relu(0.2)
    atomicMax(&menc[d], enc_f32(e));
}

__global__ __launch_bounds__(256)
void gat_segsum(const int* __restrict__ src, const int* __restrict__ dst, int E,
                int Nn, const float* __restrict__ als,
                const float* __restrict__ ald,
                const unsigned* __restrict__ menc, float* __restrict__ z) {
    int item = blockIdx.x * 256 + threadIdx.x;
    if (item >= E + Nn) return;
    int s, d; gat_item(item, E, src, dst, s, d);
    float e = als[s] + ald[d];
    e = (e > 0.0f) ? e : 0.2f * e;
    atomicAdd(&z[d], expf(e - dec_f32(menc[d])));
}

// out[d] += (w / z[d]) * h[s]; one wave per item, lane covers 8 features.
__global__ __launch_bounds__(256)
void gat_aggregate(const int* __restrict__ src, const int* __restrict__ dst,
                   int E, int Nn, const float* __restrict__ als,
                   const float* __restrict__ ald,
                   const unsigned* __restrict__ menc,
                   const float* __restrict__ z, const float* __restrict__ h,
                   float* __restrict__ out) {
    int item = blockIdx.x * 8 + (threadIdx.x >> 5);
    int lane = threadIdx.x & 31;
    if (item >= E + Nn) return;
    int s, d; gat_item(item, E, src, dst, s, d);
    float e = als[s] + ald[d];
    e = (e > 0.0f) ? e : 0.2f * e;
    float coef = expf(e - dec_f32(menc[d])) / z[d];
    const float* hp = h + (size_t)s * 256 + lane * 8;
    float* op = out + (size_t)d * 256 + lane * 8;
    #pragma unroll
    for (int i = 0; i < 8; ++i) atomicAdd(&op[i], coef * hp[i]);
}

__global__ void add_bias_vec(float* __restrict__ x, const float* __restrict__ b,
                             size_t total) {
    size_t i = (size_t)blockIdx.x * 256 + threadIdx.x;
    if (i < total) x[i] += b[i & 255];
}

// SAGE mean-aggregation scatter: one wave per edge.
__global__ __launch_bounds__(256)
void sage_scatter(const int* __restrict__ src, const int* __restrict__ dst,
                  int E, const float* __restrict__ h, float* __restrict__ ssum,
                  float* __restrict__ cnt) {
    int e = blockIdx.x * 8 + (threadIdx.x >> 5);
    int lane = threadIdx.x & 31;
    if (e >= E) return;
    int s = src[e], d = dst[e];
    const float* hp = h + (size_t)s * 256 + lane * 8;
    float* op = ssum + (size_t)d * 256 + lane * 8;
    #pragma unroll
    for (int i = 0; i < 8; ++i) atomicAdd(&op[i], hp[i]);
    if (lane == 0) atomicAdd(&cnt[d], 1.0f);
}

__global__ void sage_mean(float* __restrict__ ssum, const float* __restrict__ cnt,
                          size_t total) {
    size_t i = (size_t)blockIdx.x * 256 + threadIdx.x;
    if (i < total) ssum[i] /= fmaxf(cnt[i >> 8], 1.0f);
}

// Row L2 normalize in place; one wave per row.
__global__ __launch_bounds__(256)
void rownorm(float* __restrict__ x, int n) {
    int row  = blockIdx.x * 8 + (threadIdx.x >> 5);
    int lane = threadIdx.x & 31;
    if (row >= n) return;
    float* p = x + (size_t)row * 256 + lane * 8;
    float v[8], ss = 0.0f;
    #pragma unroll
    for (int i = 0; i < 8; ++i) { v[i] = p[i]; ss += v[i] * v[i]; }
    #pragma unroll
    for (int m = 16; m > 0; m >>= 1) ss += __shfl_xor(ss, m, 32);
    float inv = 1.0f / fmaxf(sqrtf(ss), 1e-12f);
    #pragma unroll
    for (int i = 0; i < 8; ++i) p[i] = v[i] * inv;
}

// ---------------------------------------------------------------------------
// Host-side orchestration
// ---------------------------------------------------------------------------
extern "C" void kernel_launch(void* const* d_in, const int* in_sizes, int n_in,
                              void* d_out, int out_size, void* d_ws, size_t ws_size,
                              hipStream_t stream) {
    const float* x       = (const float*)d_in[0];
    const float* W_gat   = (const float*)d_in[1];
    const float* att_src = (const float*)d_in[2];
    const float* att_dst = (const float*)d_in[3];
    const float* b_gat   = (const float*)d_in[4];
    const float* Wl1     = (const float*)d_in[8];
    const float* bl1     = (const float*)d_in[9];
    const float* Wr1     = (const float*)d_in[10];
    const int*   src1    = (const int*)d_in[13];
    const int*   dst1    = (const int*)d_in[14];

    const int H  = in_sizes[2];            // 256
    const int N0 = in_sizes[0] / H;        // 150000
    const int E1 = in_sizes[13];           // 300000
    const int n2 = out_size / H;           // 20000

    // Workspace layout (all offsets multiples of 16 bytes)
    float* h    = (float*)d_ws;                      // N0*H  (GAT pre-agg)
    float* als  = h + (size_t)N0 * H;                // N0
    float* ald  = als + N0;                          // N0
    float* zreg = ald + N0;                          // zeroed region start:
    float*    hagg = zreg;                           //   N0*H (GAT output)
    unsigned* menc = (unsigned*)(hagg + (size_t)N0 * H); // N0
    float*    z    = (float*)(menc + N0);            //   N0
    float*    ssum = z + N0;                         //   n2*H
    float*    cnt  = ssum + (size_t)n2 * H;          //   n2
    uint16_t* Bt0  = (uint16_t*)(cnt + n2);          // H*H bf16 (W_gat^T)
    uint16_t* Bt1  = Bt0 + (size_t)H * H;            // H*H bf16 (Wl1^T)
    uint16_t* Bt2  = Bt1 + (size_t)H * H;            // H*H bf16 (Wr1^T)
    float* out2 = (float*)d_out;                     // n2*H (final)

    const size_t zcount = (size_t)N0 * H + 2 * (size_t)N0 + (size_t)n2 * H + n2;
    const int items = E1 + N0;
    const int wgrid = (H * H + 255) / 256;

    // 0) zero accumulators (menc=0 == encoded -inf sentinel); prep bf16 weights
    zero_f32<<<(unsigned)((zcount + 255) / 256), 256, 0, stream>>>(zreg, zcount);
    to_bf16_t<<<wgrid, 256, 0, stream>>>(W_gat, Bt0, H, H);
    to_bf16_t<<<wgrid, 256, 0, stream>>>(Wl1,   Bt1, H, H);
    to_bf16_t<<<wgrid, 256, 0, stream>>>(Wr1,   Bt2, H, H);

    // 1) h = x @ W_gat   (bf16 WMMA, fp32 accumulate)
    gemm_dual_bf16_wmma<<<dim3((N0 + 127) / 128, H / 64), 256, 0, stream>>>(
        x, Bt0, nullptr, nullptr, nullptr, h, N0, H, H);

    // 2) attention logits per node
    rowdots<<<(N0 + 7) / 8, 256, 0, stream>>>(h, att_src, att_dst, als, ald, N0);

    // 3) softmax over incoming edges + self loops
    gat_segmax<<<(items + 255) / 256, 256, 0, stream>>>(src1, dst1, E1, N0,
                                                        als, ald, menc);
    gat_segsum<<<(items + 255) / 256, 256, 0, stream>>>(src1, dst1, E1, N0,
                                                        als, ald, menc, z);
    gat_aggregate<<<(items + 7) / 8, 256, 0, stream>>>(src1, dst1, E1, N0,
                                                       als, ald, menc, z, h, hagg);

    // 4) + GAT bias  -> hagg == h1
    add_bias_vec<<<(unsigned)(((size_t)N0 * H + 255) / 256), 256, 0, stream>>>(
        hagg, b_gat, (size_t)N0 * H);

    // 5) SAGE mean aggregation over (src1 -> dst1), n2 targets
    sage_scatter<<<(E1 + 7) / 8, 256, 0, stream>>>(src1, dst1, E1, hagg, ssum, cnt);
    sage_mean<<<(unsigned)(((size_t)n2 * H + 255) / 256), 256, 0, stream>>>(
        ssum, cnt, (size_t)n2 * H);

    // 6) out2 = mean @ Wl1 + h1[:n2] @ Wr1 + bl1   (fused dual WMMA GEMM)
    gemm_dual_bf16_wmma<<<dim3((n2 + 127) / 128, H / 64), 256, 0, stream>>>(
        ssum, Bt1, hagg, Bt2, bl1, out2, n2, H, H);

    // 7) row-wise L2 normalize in place
    rownorm<<<(n2 + 7) / 8, 256, 0, stream>>>(out2, n2);
}
